// PointNet_21483426414945
// MI455X (gfx1250) — matrix operand order repaired and checked by reading, not compiled, not run on hardware
//
#include <hip/hip_runtime.h>
#include <math.h>

// ---------------------------------------------------------------------------
// PointNet++ (SA x2 + global SA + head) for MI455X / gfx1250.
// GEMMs: v_wmma_f32_16x16x32_f16 (wave32), fp16 activations, fp32 accum.
// LDS: A tile row-major (lane-contiguous frags -> ds_load_b128), W tile
// transposed [col][k] so B frags are also lane-contiguous -> ds_load_b128.
// K staged 64 per iteration -> 2 wmma per stage. Per-batch chunking keeps
// ping-pong activation buffers (<=16MB) resident in the 192MB L2.
// ---------------------------------------------------------------------------

typedef _Float16 half_t;
typedef _Float16 v16h __attribute__((ext_vector_type(16)));
typedef _Float16 v8h  __attribute__((ext_vector_type(8)));
typedef float    v8f  __attribute__((ext_vector_type(8)));

// ---- WMMA fragment loads from LDS, per CDNA5 ISA 7.12.2 layouts (wave32) ----
// A tile: row-major [16][64]. For k-substep base `ks`, lane (hs=lane>>4,
// r=lane&15) needs K = ks + hs*8 + 0..7 and K = ks + 16 + hs*8 + 0..7:
// two contiguous 8-half runs -> 2x ds_load_b128.
__device__ __forceinline__ void load_a_frag(const half_t* sA, int r, int hs, int ks, v16h& a) {
#pragma unroll
  for (int h = 0; h < 16; ++h) {
    int v = h >> 1, slot = h & 1;
    int k = (v < 4) ? (hs * 8 + 2 * v + slot) : (16 + hs * 8 + 2 * (v - 4) + slot);
    a[h] = sA[r * 64 + ks + k];
  }
}
// B tile: transposed [64 cols][64 k]. Lane (hs, n=lane&15) at colBase needs
// K = ks + hs*8 + 0..7 and K = ks + 16 + hs*8 + 0..7 for col colBase+n.
__device__ __forceinline__ void load_b_frag(const half_t* sBt, int colBase, int n, int hs,
                                            int ks, v16h& b) {
  const half_t* p = sBt + (colBase + n) * 64 + ks + hs * 8;
#pragma unroll
  for (int h = 0; h < 16; ++h) b[h] = p[(h < 8) ? h : h + 8];
}

// ---- Fused GEMM + bias + optional ReLU: Y[R,N] = act(X[R,Kd] @ W[Kd,N] + b)
// Block = 128 threads (4 waves); block tile 16 rows x 64 cols; each wave owns
// one 16x16 tile. Kd multiple of 64, R multiple of 16, N multiple of 16.
__global__ void gemm_bias_act(const half_t* __restrict__ X, const half_t* __restrict__ W,
                              const float* __restrict__ bias, half_t* __restrict__ Y,
                              int R, int Kd, int N, int relu)
{
  __shared__ half_t sA[16 * 64];     // 2 KB
  __shared__ half_t sBt[64 * 64];    // 8 KB, transposed [col][k]
  int tid = threadIdx.x;
  int wave = tid >> 5;
  int lane = tid & 31;
  int hs = (lane >> 4) & 1;
  int rl = lane & 15;                 // A row / B col within tile
  int rowBase = blockIdx.x * 16;
  int colBlock = blockIdx.y * 64;
  int myCol = colBlock + wave * 16;   // wave-uniform
  v8f acc = {};
  for (int k0 = 0; k0 < Kd; k0 += 64) {
    // Stage A: 16x64 halves, one aligned 16B vector per thread (16*8 = 128).
    {
      int r = tid >> 3;
      int c = (tid & 7) * 8;
      *(v8h*)(sA + r * 64 + c) =
          *(const v8h*)(X + (size_t)(rowBase + r) * Kd + (k0 + c));
    }
    // Prefetch next A k-tile (emits global_prefetch_b8).
    if (k0 + 64 < Kd)
      __builtin_prefetch(X + (size_t)(rowBase + (tid >> 3)) * Kd + (k0 + 64), 0, 1);
    // Stage W transposed: 64 cols x 64 k. Per idx: 8 coalesced global loads
    // (consecutive lanes -> consecutive cols) + one aligned 16B LDS store.
    for (int idx = tid; idx < 512; idx += 128) {
      int col = idx & 63;
      int kg  = (idx >> 6) * 8;
      int gcol = colBlock + col;
      v8h tmp;
#pragma unroll
      for (int j = 0; j < 8; ++j)
        tmp[j] = (gcol < N) ? W[(size_t)(k0 + kg + j) * N + gcol] : (half_t)0.f;
      *(v8h*)(sBt + col * 64 + kg) = tmp;
    }
    __syncthreads();
    if (myCol < N) {                  // wave-uniform: EXEC stays all-1s
#pragma unroll
      for (int ks = 0; ks < 64; ks += 32) {
        v16h a, b;
        load_a_frag(sA, rl, hs, ks, a);
        load_b_frag(sBt, wave * 16, rl, hs, ks, b);
        acc = __builtin_amdgcn_wmma_f32_16x16x32_f16(false, a, false, b,
                                                     (short)0, acc, false, false);
      }
    }
    __syncthreads();
  }
  if (myCol < N) {
    float bv = bias ? bias[myCol + rl] : 0.f;
#pragma unroll
    for (int v = 0; v < 8; ++v) {     // C/D layout: M = v + hs*8, N = lane%16
      float val = acc[v] + bv;
      if (relu) val = fmaxf(val, 0.f);
      int row = rowBase + hs * 8 + v;
      Y[(size_t)row * N + (myCol + rl)] = (half_t)val;
    }
  }
}

// ---- fp32 -> fp16 weight conversion with zero padding (K->x64, N->x16) ----
__global__ void convert_weight(const float* __restrict__ W, const float* __restrict__ b,
                               half_t* __restrict__ Wd, float* __restrict__ bd,
                               int Cin, int Cout, int CinP, int CoutP)
{
  int i = blockIdx.x * blockDim.x + threadIdx.x;
  int total = CinP * CoutP;
  if (i < total) {
    int r = i / CoutP, c = i % CoutP;
    float v = (r < Cin && c < Cout) ? W[r * Cout + c] : 0.f;
    Wd[i] = (half_t)v;
  }
  if (i < CoutP) bd[i] = (i < Cout) ? b[i] : 0.f;
}

// ---- Farthest point sampling: one block per cloud; points + mind in LDS ----
__global__ void fps_kernel(const float* __restrict__ pos, int N, int M,
                           int* __restrict__ idx_out, float* __restrict__ posq_out)
{
  __shared__ float px[2048], py[2048], pz[2048], mind[2048];
  __shared__ float rv[256];
  __shared__ int   ri[256];
  __shared__ int   s_last;
  int b = blockIdx.x, t = threadIdx.x;
  const float* P = pos + (size_t)b * N * 3;
  for (int i = t; i < N; i += 256) {
    px[i] = P[i * 3 + 0]; py[i] = P[i * 3 + 1]; pz[i] = P[i * 3 + 2];
    mind[i] = 3.4e38f;
  }
  if (t == 0) { s_last = 0; idx_out[(size_t)b * M + 0] = 0; }
  __syncthreads();
  for (int s = 1; s < M; ++s) {
    int last = s_last;
    float lx = px[last], ly = py[last], lz = pz[last];
    float bestv = -1.f; int besti = 0x7fffffff;
    for (int i = t; i < N; i += 256) {
      float dx = px[i] - lx, dy = py[i] - ly, dz = pz[i] - lz;
      float d = dx * dx + dy * dy + dz * dz;
      float m = fminf(mind[i], d);
      mind[i] = m;
      if (m > bestv) { bestv = m; besti = i; }
    }
    rv[t] = bestv; ri[t] = besti;
    __syncthreads();
    for (int off = 128; off > 0; off >>= 1) {
      if (t < off) {
        if (rv[t + off] > rv[t] || (rv[t + off] == rv[t] && ri[t + off] < ri[t])) {
          rv[t] = rv[t + off]; ri[t] = ri[t + off];
        }
      }
      __syncthreads();
    }
    if (t == 0) { s_last = ri[0]; idx_out[(size_t)b * M + s] = ri[0]; }
    __syncthreads();
  }
  for (int s = t; s < M; s += 256) {
    int j = idx_out[(size_t)b * M + s];
    posq_out[((size_t)b * M + s) * 3 + 0] = px[j];
    posq_out[((size_t)b * M + s) * 3 + 1] = py[j];
    posq_out[((size_t)b * M + s) * 3 + 2] = pz[j];
  }
}

// ---- Ball query: one wave32 per query; deterministic ballot+prefix append ----
__global__ void ball_query(const float* __restrict__ pos, const float* __restrict__ posq,
                           int N, int M, int nQuery, float r2, int* __restrict__ nbr)
{
  int gw = (blockIdx.x * blockDim.x + threadIdx.x) >> 5;
  int lane = threadIdx.x & 31;
  if (gw >= nQuery) return;                   // wave-uniform
  int b = gw / M;
  const float* P = pos + (size_t)b * N * 3;
  float qx = posq[(size_t)gw * 3 + 0];
  float qy = posq[(size_t)gw * 3 + 1];
  float qz = posq[(size_t)gw * 3 + 2];
  int* out = nbr + (size_t)gw * 64;
  out[lane] = -1; out[lane + 32] = -1;
  int count = 0;
  for (int base = 0; base < N && count < 64; base += 32) {
    int j = base + lane;
    float dx = P[j * 3 + 0] - qx, dy = P[j * 3 + 1] - qy, dz = P[j * 3 + 2] - qz;
    bool pred = (dx * dx + dy * dy + dz * dz) <= r2;
    unsigned m32 = (unsigned)__ballot(pred);
    int prefix = __popc(m32 & ((1u << lane) - 1u));
    int p = count + prefix;
    if (pred && p < 64) out[p] = j;
    count += __popc(m32);
  }
}

// ---- Row gathers (widths padded to x64 with zeros) ----
__global__ void gather_sa1(const float* __restrict__ x, const float* __restrict__ pos,
                           const float* __restrict__ posq, const int* __restrict__ nbr,
                           int b, int N, int M, half_t* __restrict__ Xrows)
{
  int r = blockIdx.x * blockDim.x + threadIdx.x;
  if (r >= M * 64) return;
  int m = r >> 6;
  int j = nbr[((size_t)b * M + m) * 64 + (r & 63)];
  half_t* row = Xrows + (size_t)r * 64;
  float qx = posq[((size_t)b * M + m) * 3 + 0];
  float qy = posq[((size_t)b * M + m) * 3 + 1];
  float qz = posq[((size_t)b * M + m) * 3 + 2];
  if (j >= 0) {
    const float* xv = x + ((size_t)b * N + j) * 3;
    const float* pv = pos + ((size_t)b * N + j) * 3;
    row[0] = (half_t)xv[0]; row[1] = (half_t)xv[1]; row[2] = (half_t)xv[2];
    row[3] = (half_t)(pv[0] - qx); row[4] = (half_t)(pv[1] - qy); row[5] = (half_t)(pv[2] - qz);
  } else {
    for (int c = 0; c < 6; ++c) row[c] = (half_t)0.f;
  }
  for (int c = 6; c < 64; ++c) row[c] = (half_t)0.f;
}

__global__ void gather_sa2(const half_t* __restrict__ x1, const float* __restrict__ pos1,
                           const float* __restrict__ pos2, const int* __restrict__ nbr,
                           int b, half_t* __restrict__ Xrows)
{
  int i = blockIdx.x * blockDim.x + threadIdx.x;   // over 16384*192
  if (i >= 16384 * 192) return;
  int r = i / 192, c = i % 192;
  int m = r >> 6;
  int j = nbr[((size_t)b * 256 + m) * 64 + (r & 63)];
  half_t v = (half_t)0.f;
  if (j >= 0) {
    if (c < 128)      v = x1[((size_t)b * 1024 + j) * 128 + c];
    else if (c < 131) v = (half_t)(pos1[((size_t)b * 1024 + j) * 3 + (c - 128)] -
                                   pos2[((size_t)b * 256 + m) * 3 + (c - 128)]);
  }
  Xrows[i] = v;
}

__global__ void gather_global(const half_t* __restrict__ x2, const float* __restrict__ pos2,
                              half_t* __restrict__ Xrows)
{
  int i = blockIdx.x * blockDim.x + threadIdx.x;   // over 2048*320
  if (i >= 2048 * 320) return;
  int r = i / 320, c = i % 320;
  half_t v = (half_t)0.f;
  if (c < 256)      v = x2[(size_t)r * 256 + c];
  else if (c < 259) v = (half_t)pos2[(size_t)r * 3 + (c - 256)];
  Xrows[i] = v;
}

// ---- Masked max over the K=64 neighbor axis ----
__global__ void max_over_k(const half_t* __restrict__ Yrows, const int* __restrict__ nbr,
                           int b, int M, int C, half_t* __restrict__ out)
{
  int i = blockIdx.x * blockDim.x + threadIdx.x;
  if (i >= M * C) return;
  int m = i / C, c = i % C;
  const int* nb = nbr + ((size_t)b * M + m) * 64;
  float best = -3.4e38f;
  for (int k = 0; k < 64; ++k)
    if (nb[k] >= 0) best = fmaxf(best, (float)Yrows[((size_t)(m * 64 + k)) * C + c]);
  out[((size_t)b * M + m) * C + c] = (half_t)best;
}

// ---- Global max pool over 256 points; rows 8..15 zero-padded for the head ----
__global__ void global_max(const half_t* __restrict__ H, half_t* __restrict__ g)
{
  int i = blockIdx.x * blockDim.x + threadIdx.x;   // over 16*1024
  if (i >= 16 * 1024) return;
  int b = i >> 10, c = i & 1023;
  float best = 0.f;
  if (b < 8) {
    best = -3.4e38f;
    for (int m = 0; m < 256; ++m)
      best = fmaxf(best, (float)H[((size_t)b * 256 + m) * 1024 + c]);
  }
  g[i] = (half_t)best;
}

// ---- log_softmax over 10 logits ----
__global__ void logsoftmax_kernel(const half_t* __restrict__ Z, float* __restrict__ out)
{
  int b = blockIdx.x, t = threadIdx.x;
  __shared__ float v[10];
  if (t < 10) v[t] = (float)Z[b * 16 + t];
  __syncthreads();
  if (t == 0) {
    float mx = v[0];
    for (int i = 1; i < 10; ++i) mx = fmaxf(mx, v[i]);
    float s = 0.f;
    for (int i = 0; i < 10; ++i) s += expf(v[i] - mx);
    float lse = mx + logf(s);
    for (int i = 0; i < 10; ++i) out[b * 10 + i] = v[i] - lse;
  }
}

// ---------------------------------------------------------------------------
extern "C" void kernel_launch(void* const* d_in, const int* in_sizes, int n_in,
                              void* d_out, int out_size, void* d_ws, size_t ws_size,
                              hipStream_t stream)
{
  (void)in_sizes; (void)n_in; (void)out_size; (void)ws_size;
  const float* x   = (const float*)d_in[0];
  const float* pos = (const float*)d_in[1];

  struct LayerDim { int cin, cout, cinP, coutP; };
  static const LayerDim L[12] = {
    {6, 64, 64, 64},     {64, 64, 64, 64},    {64, 128, 64, 128},
    {131, 128, 192, 128},{128, 128, 128, 128},{128, 256, 128, 256},
    {259, 256, 320, 256},{256, 512, 256, 512},{512, 1024, 512, 1024},
    {1024, 512, 1024, 512},{512, 256, 512, 256},{256, 10, 256, 16}
  };

  char* base = (char*)d_ws;
  size_t off = 0;
  auto carve = [&](size_t bytes) -> char* {
    char* p = base + off;
    off = (off + bytes + 255) & ~(size_t)255;
    return p;
  };

  half_t* Wh[12]; float* Bh[12];
  for (int l = 0; l < 12; ++l) {
    Wh[l] = (half_t*)carve((size_t)L[l].cinP * L[l].coutP * sizeof(half_t));
    Bh[l] = (float*)carve((size_t)L[l].coutP * sizeof(float));
  }
  int*    idx1 = (int*)carve((size_t)8 * 1024 * 4);
  float*  pos1 = (float*)carve((size_t)8 * 1024 * 3 * 4);
  int*    idx2 = (int*)carve((size_t)8 * 256 * 4);
  float*  pos2 = (float*)carve((size_t)8 * 256 * 3 * 4);
  int*    nbr1 = (int*)carve((size_t)8 * 1024 * 64 * 4);
  int*    nbr2 = (int*)carve((size_t)8 * 256 * 64 * 4);
  half_t* x1   = (half_t*)carve((size_t)8 * 1024 * 128 * sizeof(half_t));
  half_t* x2   = (half_t*)carve((size_t)8 * 256 * 256 * sizeof(half_t));
  half_t* g    = (half_t*)carve((size_t)16 * 1024 * sizeof(half_t));
  const size_t BUFH = (size_t)65536 * 128;           // max chunk activation size
  half_t* bufA = (half_t*)carve(BUFH * sizeof(half_t));
  half_t* bufB = (half_t*)carve(BUFH * sizeof(half_t));

  // Weight conversion / padding
  for (int l = 0; l < 12; ++l) {
    const float* Ws = (const float*)d_in[2 + 2 * l];
    const float* bs = (const float*)d_in[3 + 2 * l];
    int tot = L[l].cinP * L[l].coutP;
    convert_weight<<<(tot + 255) / 256, 256, 0, stream>>>(
        Ws, bs, Wh[l], Bh[l], L[l].cin, L[l].cout, L[l].cinP, L[l].coutP);
  }

  // ---- SA1: N=2048 -> M=1024, r=0.2 ----
  fps_kernel<<<8, 256, 0, stream>>>(pos, 2048, 1024, idx1, pos1);
  ball_query<<<(8 * 1024 * 32 + 255) / 256, 256, 0, stream>>>(
      pos, pos1, 2048, 1024, 8 * 1024, 0.04f, nbr1);
  for (int b = 0; b < 8; ++b) {
    gather_sa1<<<(65536 + 255) / 256, 256, 0, stream>>>(x, pos, pos1, nbr1, b, 2048, 1024, bufA);
    gemm_bias_act<<<dim3(65536 / 16, 1), 128, 0, stream>>>(bufA, Wh[0], Bh[0], bufB, 65536, 64, 64, 1);
    gemm_bias_act<<<dim3(65536 / 16, 1), 128, 0, stream>>>(bufB, Wh[1], Bh[1], bufA, 65536, 64, 64, 1);
    gemm_bias_act<<<dim3(65536 / 16, 2), 128, 0, stream>>>(bufA, Wh[2], Bh[2], bufB, 65536, 64, 128, 0);
    max_over_k<<<(1024 * 128 + 255) / 256, 256, 0, stream>>>(bufB, nbr1, b, 1024, 128, x1);
  }

  // ---- SA2: N=1024 -> M=256, r=0.4 ----
  fps_kernel<<<8, 256, 0, stream>>>(pos1, 1024, 256, idx2, pos2);
  ball_query<<<(8 * 256 * 32 + 255) / 256, 256, 0, stream>>>(
      pos1, pos2, 1024, 256, 8 * 256, 0.16f, nbr2);
  for (int b = 0; b < 8; ++b) {
    gather_sa2<<<(16384 * 192 + 255) / 256, 256, 0, stream>>>(x1, pos1, pos2, nbr2, b, bufA);
    gemm_bias_act<<<dim3(16384 / 16, 2), 128, 0, stream>>>(bufA, Wh[3], Bh[3], bufB, 16384, 192, 128, 1);
    gemm_bias_act<<<dim3(16384 / 16, 2), 128, 0, stream>>>(bufB, Wh[4], Bh[4], bufA, 16384, 128, 128, 1);
    gemm_bias_act<<<dim3(16384 / 16, 4), 128, 0, stream>>>(bufA, Wh[5], Bh[5], bufB, 16384, 128, 256, 0);
    max_over_k<<<(256 * 256 + 255) / 256, 256, 0, stream>>>(bufB, nbr2, b, 256, 256, x2);
  }

  // ---- Global SA ----
  gather_global<<<(2048 * 320 + 255) / 256, 256, 0, stream>>>(x2, pos2, bufA);
  gemm_bias_act<<<dim3(2048 / 16, 4), 128, 0, stream>>>(bufA, Wh[6], Bh[6], bufB, 2048, 320, 256, 1);
  gemm_bias_act<<<dim3(2048 / 16, 8), 128, 0, stream>>>(bufB, Wh[7], Bh[7], bufA, 2048, 256, 512, 1);
  gemm_bias_act<<<dim3(2048 / 16, 16), 128, 0, stream>>>(bufA, Wh[8], Bh[8], bufB, 2048, 512, 1024, 0);
  global_max<<<(16 * 1024 + 255) / 256, 256, 0, stream>>>(bufB, g);

  // ---- Head MLP (rows padded 8 -> 16) ----
  gemm_bias_act<<<dim3(1, 8), 128, 0, stream>>>(g,    Wh[9],  Bh[9],  bufA, 16, 1024, 512, 1);
  gemm_bias_act<<<dim3(1, 4), 128, 0, stream>>>(bufA, Wh[10], Bh[10], bufB, 16, 512, 256, 1);
  gemm_bias_act<<<dim3(1, 1), 128, 0, stream>>>(bufB, Wh[11], Bh[11], bufA, 16, 256, 16, 0);

  logsoftmax_kernel<<<8, 32, 0, stream>>>(bufA, (float*)d_out);
}
